// DynamicConv_274877907016
// MI455X (gfx1250) — matrix operand order repaired and checked by reading, not compiled, not run on hardware
//
#include <hip/hip_runtime.h>
#include <hip/hip_bf16.h>
#include <math.h>

#define B_ 8
#define R_ 4
#define C_ 64
#define H_ 128
#define W_ 128
#define HW_ (H_*W_)
#define C2_ 128
#define E_ 8
#define RF_ 256

typedef __attribute__((ext_vector_type(16))) __bf16 v16bf;
typedef __attribute__((ext_vector_type(8)))  float  v8f;

union Frag  { v16bf v; uint4 q[2]; unsigned short s[16]; };
union Pack8 { uint4 q; unsigned short s[8]; };

__device__ __forceinline__ unsigned short f2bf(float f) {
  union { float f; unsigned u; } v; v.f = f;
  unsigned r = v.u + 0x7FFFu + ((v.u >> 16) & 1u);  // round-to-nearest-even
  return (unsigned short)(r >> 16);
}
__device__ __forceinline__ float bf2f(unsigned short s) {
  union { unsigned u; float f; } v; v.u = ((unsigned)s) << 16;
  return v.f;
}

// ---- NCHW f32 -> NHWC bf16 ----------------------------------------------
__global__ void k_cvt(const float* __restrict__ x, unsigned short* __restrict__ xbf) {
  int idx = blockIdx.x * 256 + threadIdx.x;
  if (idx >= B_*HW_*C_) return;
  int c = idx % C_;
  int p = (idx / C_) % HW_;
  int b = idx / (C_ * HW_);
  xbf[idx] = f2bf(x[((size_t)(b*C_ + c))*HW_ + p]);
}

// ---- routing weights: sigmoid(routing @ W + b), stored [R][B][E] ---------
__global__ void k_rw(const float* __restrict__ routing, const float* __restrict__ rW,
                     const float* __restrict__ rb, float* __restrict__ rw) {
  int t = threadIdx.x;            // 256 = B*R*E
  int b = t >> 5, r = (t >> 3) & 3, e = t & 7;
  float s = rb[e];
  for (int k = 0; k < RF_; ++k)
    s += routing[(b*R_ + r)*RF_ + k] * rW[k*E_ + e];
  rw[(r*B_ + b)*E_ + e] = 1.f / (1.f + expf(-s));
}

// ---- mix expert kernels W1: out [b][tap][C2][C] bf16 ---------------------
__global__ void k_mixw1(const float* __restrict__ rw, const float* __restrict__ W1,
                        unsigned short* __restrict__ w1m, int r) {
  int idx = blockIdx.x * 256 + threadIdx.x;
  if (idx >= B_*9*C2_*C_) return;
  int c   = idx % C_;
  int o   = (idx / C_) % C2_;
  int tap = (idx / (C_*C2_)) % 9;
  int b   = idx / (C_*C2_*9);
  float s = 0.f;
#pragma unroll
  for (int e = 0; e < E_; ++e)
    s += rw[(r*B_ + b)*E_ + e] * W1[(size_t)((e*C2_ + o)*C_ + c)*9 + tap];
  w1m[idx] = f2bf(s);
}

// ---- mix expert kernels W2: out [b][tap][C][C2] bf16 ---------------------
__global__ void k_mixw2(const float* __restrict__ srw, const float* __restrict__ W2,
                        unsigned short* __restrict__ w2m) {
  int idx = blockIdx.x * 256 + threadIdx.x;
  if (idx >= B_*9*C_*C2_) return;
  int c2  = idx % C2_;
  int o   = (idx / C2_) % C_;
  int tap = (idx / (C2_*C_)) % 9;
  int b   = idx / (C2_*C_*9);
  float s = 0.f;
#pragma unroll
  for (int e = 0; e < E_; ++e)
    s += srw[b*E_ + e] * W2[(size_t)((e*C_ + o)*C2_ + c2)*9 + tap];
  w2m[idx] = f2bf(s);
}

// ---- conv1: implicit GEMM, M=C2(8 tiles) N=16 pixels K=C*9, bf16 WMMA ----
// block = one (b,h) row, 8 waves x 16 pixels. Writes xc NHWC bf16 + PReLU.
__global__ void k_conv1(const unsigned short* __restrict__ xbf,
                        const unsigned short* __restrict__ w1m,
                        unsigned short* __restrict__ xc,
                        const float* __restrict__ prelu_a) {
  const int wave = threadIdx.x >> 5;
  const int lane = threadIdx.x & 31;
  const int hi   = lane >> 4;
  const int n    = lane & 15;
  const int b    = blockIdx.x >> 7;
  const int h    = blockIdx.x & 127;
  const int w0   = wave * 16;

  v8f acc[8] = {};

  for (int tap = 0; tap < 9; ++tap) {
    const int kh = tap / 3, kw = tap % 3;
    const int hh = h + kh - 1;
    const int ww = w0 + n + kw - 1;
    const bool oob = ((unsigned)hh >= (unsigned)H_) || ((unsigned)ww >= (unsigned)W_);
    const unsigned short* bbase = xbf + (size_t)(b*HW_ + hh*W_ + ww) * C_;
    const unsigned short* abase = w1m + (size_t)((b*9 + tap)*C2_ + n) * C_;
#pragma unroll
    for (int s = 0; s < 2; ++s) {            // K = 64 channels = 2 x 32
      Frag bf;
      if (oob) {
        bf.q[0] = make_uint4(0u,0u,0u,0u);
        bf.q[1] = make_uint4(0u,0u,0u,0u);
      } else {
        const unsigned short* p = bbase + s*32 + hi*8;
        bf.q[0] = *(const uint4*)(p);
        bf.q[1] = *(const uint4*)(p + 16);
      }
#pragma unroll
      for (int mt = 0; mt < 8; ++mt) {
        Frag af;
        const unsigned short* p = abase + (size_t)mt*16*C_ + s*32 + hi*8;
        af.q[0] = *(const uint4*)(p);
        af.q[1] = *(const uint4*)(p + 16);
        acc[mt] = __builtin_amdgcn_wmma_f32_16x16x32_bf16(
            false, af.v, false, bf.v, (short)0, acc[mt], false, false);
      }
    }
  }

  const float a = prelu_a[0];
  const int p = h*W_ + w0 + n;
  unsigned short* dst0 = xc + (size_t)(b*HW_ + p)*C2_ + hi*8;
#pragma unroll
  for (int mt = 0; mt < 8; ++mt) {
    Pack8 pk;
#pragma unroll
    for (int j = 0; j < 8; ++j) {
      float v = acc[mt][j];
      v = (v >= 0.f) ? v : a*v;
      pk.s[j] = f2bf(v);
    }
    *(uint4*)(dst0 + mt*16) = pk.q;
  }
}

// ---- global average pool over HW per (b, channel) ------------------------
__global__ void k_pool(const unsigned short* __restrict__ xc, float* __restrict__ pooled) {
  __shared__ float sm[256];
  const int b = blockIdx.x >> 7;       // C2_ = 128 blocks per batch
  const int o = blockIdx.x & 127;
  float s = 0.f;
  for (int p = threadIdx.x; p < HW_; p += 256)
    s += bf2f(xc[(size_t)(b*HW_ + p)*C2_ + o]);
  sm[threadIdx.x] = s;
  __syncthreads();
  for (int st = 128; st > 0; st >>= 1) {
    if ((int)threadIdx.x < st) sm[threadIdx.x] += sm[threadIdx.x + st];
    __syncthreads();
  }
  if (threadIdx.x == 0) pooled[b*C2_ + o] = sm[0] * (1.0f / HW_);
}

// ---- self-routing linear (no sigmoid): srw = pooled @ sr_W + sr_b --------
__global__ void k_srw(const float* __restrict__ pooled, const float* __restrict__ srW,
                      const float* __restrict__ srb, float* __restrict__ srw) {
  int t = threadIdx.x;
  if (t >= B_*E_) return;
  int b = t >> 3, e = t & 7;
  float s = srb[e];
  for (int c = 0; c < C2_; ++c)
    s += pooled[b*C2_ + c] * srW[c*E_ + e];
  srw[t] = s;
}

// ---- conv2 + PReLU + fused 1x1 Wout reduction + bout, writes d_out -------
__global__ void k_conv2(const unsigned short* __restrict__ xc,
                        const unsigned short* __restrict__ w2m,
                        const float* __restrict__ Wout,
                        const float* __restrict__ bout,
                        const float* __restrict__ prelu_a,
                        float* __restrict__ out, int r) {
  const int wave = threadIdx.x >> 5;
  const int lane = threadIdx.x & 31;
  const int hi   = lane >> 4;
  const int n    = lane & 15;
  const int b    = blockIdx.x >> 7;
  const int h    = blockIdx.x & 127;
  const int w0   = wave * 16;

  v8f acc[4] = {};

  for (int tap = 0; tap < 9; ++tap) {
    const int kh = tap / 3, kw = tap % 3;
    const int hh = h + kh - 1;
    const int ww = w0 + n + kw - 1;
    const bool oob = ((unsigned)hh >= (unsigned)H_) || ((unsigned)ww >= (unsigned)W_);
    const unsigned short* bbase = xc  + (size_t)(b*HW_ + hh*W_ + ww) * C2_;
    const unsigned short* abase = w2m + (size_t)((b*9 + tap)*C_ + n) * C2_;
#pragma unroll
    for (int s = 0; s < 4; ++s) {            // K = 128 channels = 4 x 32
      Frag bf;
      if (oob) {
        bf.q[0] = make_uint4(0u,0u,0u,0u);
        bf.q[1] = make_uint4(0u,0u,0u,0u);
      } else {
        const unsigned short* p = bbase + s*32 + hi*8;
        bf.q[0] = *(const uint4*)(p);
        bf.q[1] = *(const uint4*)(p + 16);
      }
#pragma unroll
      for (int mt = 0; mt < 4; ++mt) {
        Frag af;
        const unsigned short* p = abase + (size_t)mt*16*C2_ + s*32 + hi*8;
        af.q[0] = *(const uint4*)(p);
        af.q[1] = *(const uint4*)(p + 16);
        acc[mt] = __builtin_amdgcn_wmma_f32_16x16x32_bf16(
            false, af.v, false, bf.v, (short)0, acc[mt], false, false);
      }
    }
  }

  const float a = prelu_a[0];
  float part = 0.f;
#pragma unroll
  for (int mt = 0; mt < 4; ++mt) {
#pragma unroll
    for (int j = 0; j < 8; ++j) {
      float v = acc[mt][j];
      v = (v >= 0.f) ? v : a*v;
      part += v * Wout[mt*16 + hi*8 + j];
    }
  }
  part += __shfl_xor(part, 16, 32);          // combine lane pair sharing a pixel
  if (hi == 0)
    out[(size_t)(b*R_ + r)*HW_ + h*W_ + w0 + n] = part + bout[0];
}

extern "C" void kernel_launch(void* const* d_in, const int* in_sizes, int n_in,
                              void* d_out, int out_size, void* d_ws, size_t ws_size,
                              hipStream_t stream) {
  (void)in_sizes; (void)n_in; (void)out_size; (void)ws_size;
  const float* x         = (const float*)d_in[0];
  const float* routing   = (const float*)d_in[1];
  const float* routing_W = (const float*)d_in[2];
  const float* routing_b = (const float*)d_in[3];
  const float* W1        = (const float*)d_in[4];
  const float* prelu_a   = (const float*)d_in[5];
  const float* W2        = (const float*)d_in[6];
  const float* sr_W      = (const float*)d_in[7];
  const float* sr_b      = (const float*)d_in[8];
  const float* Wout      = (const float*)d_in[9];
  const float* bout      = (const float*)d_in[10];
  float* out = (float*)d_out;

  char* ws = (char*)d_ws;
  size_t off = 0;
  unsigned short* xbf = (unsigned short*)(ws + off); off += (size_t)B_*HW_*C_*2;   // 16.8 MB
  unsigned short* xc  = (unsigned short*)(ws + off); off += (size_t)B_*HW_*C2_*2;  // 33.6 MB
  unsigned short* w1m = (unsigned short*)(ws + off); off += (size_t)B_*9*C2_*C_*2; // 1.18 MB
  unsigned short* w2m = (unsigned short*)(ws + off); off += (size_t)B_*9*C_*C2_*2; // 1.18 MB
  float* rw     = (float*)(ws + off); off += (size_t)R_*B_*E_*4;
  float* pooled = (float*)(ws + off); off += (size_t)B_*C2_*4;
  float* srw    = (float*)(ws + off); off += (size_t)B_*E_*4;

  k_cvt<<<(B_*HW_*C_ + 255)/256, 256, 0, stream>>>(x, xbf);
  k_rw <<<1, 256, 0, stream>>>(routing, routing_W, routing_b, rw);

  for (int r = 0; r < R_; ++r) {
    k_mixw1<<<(B_*9*C2_*C_ + 255)/256, 256, 0, stream>>>(rw, W1, w1m, r);
    k_conv1<<<B_*H_, 256, 0, stream>>>(xbf, w1m, xc, prelu_a);
    k_pool <<<B_*C2_, 256, 0, stream>>>(xc, pooled);
    k_srw  <<<1, 64, 0, stream>>>(pooled, sr_W, sr_b, srw);
    k_mixw2<<<(B_*9*C_*C2_ + 255)/256, 256, 0, stream>>>(srw, W2, w2m);
    k_conv2<<<B_*H_, 256, 0, stream>>>(xc, w2m, Wout, bout, prelu_a, out, r);
  }
}
